// MixerModel_10986526343591
// MI455X (gfx1250) — matrix operand order repaired and checked by reading, not compiled, not run on hardware
//
#include <hip/hip_runtime.h>
#include <math.h>

#define B_   8
#define T_   2048
#define DM   64
#define DI   128
#define DS_  16
#define XPN  36   // dt_rank(4) + 2*d_state(32)
#define XPL  48   // padded x-proj width (3 WMMA tiles)

typedef __attribute__((ext_vector_type(16))) _Float16 v16h;
typedef __attribute__((ext_vector_type(8)))  float    v8f;

__device__ __forceinline__ float sigmoidf_(float x) { return 1.0f / (1.0f + __expf(-x)); }

// ---------------------------------------------------------------------------
// Row-major GEMM via WMMA f16->f32:  C[M x N] = A[M x K] * W[N x K]^T
// One wave computes a 16 x (16*NBLK) strip; A fragment reused NBLK times.
// K is compile-time (fully unrolled); N = NtileGroups * 16 * NBLK exactly.
// No divergent guards anywhere: weights must be pre-padded.
// ---------------------------------------------------------------------------
template<int K, int NBLK>
__global__ void gemm_wmma(const float* __restrict__ A, const float* __restrict__ W,
                          float* __restrict__ C, int NtileGroups, int ldC)
{
    int wave = blockIdx.x * (blockDim.x >> 5) + (threadIdx.x >> 5);
    int mt = wave / NtileGroups;
    int ng = wave % NtileGroups;
    int m0 = mt * 16;
    int n0 = ng * 16 * NBLK;

    int lane = threadIdx.x & 31;
    int half = lane >> 4;                  // 0: lanes 0-15, 1: lanes 16-31
    int lr   = lane & 15;

    const float* Ap = A + (size_t)(m0 + lr) * K;
    const float* Wp[NBLK];
    #pragma unroll
    for (int u = 0; u < NBLK; ++u)
        Wp[u] = W + (size_t)(n0 + u * 16 + lr) * K;

    v8f acc[NBLK];
    #pragma unroll
    for (int u = 0; u < NBLK; ++u) acc[u] = (v8f){};

    #pragma unroll
    for (int k0 = 0; k0 < K; k0 += 32) {
        // A fragment (16x32): lane<16 -> K {k0..k0+15}; lane>=16 -> K {k0+16..k0+31}
        v16h a;
        int kb = k0 + half * 16;
        #pragma unroll
        for (int j = 0; j < 8; ++j) a[j]     = (_Float16)Ap[kb + j];
        #pragma unroll
        for (int j = 0; j < 8; ++j) a[8 + j] = (_Float16)Ap[kb + 8 + j];
        // B fragments (32x16 each): lane<16 -> K {k0..k0+15}; lane>=16 -> K {k0+16..k0+31}
        int kc = k0 + half * 16;
        #pragma unroll
        for (int u = 0; u < NBLK; ++u) {
            v16h bf;
            #pragma unroll
            for (int j = 0; j < 16; ++j) bf[j] = (_Float16)Wp[u][kc + j];
            acc[u] = __builtin_amdgcn_wmma_f32_16x16x32_f16(
                false, a, false, bf, (short)0, acc[u], false, false);
        }
    }
    #pragma unroll
    for (int u = 0; u < NBLK; ++u) {
        int col = n0 + u * 16 + lr;
        #pragma unroll
        for (int r = 0; r < 8; ++r) {
            int row = m0 + r + half * 8;   // C/D layout: VGPR r -> M=r (lanes<16) / M=8+r
            C[(size_t)row * ldC + col] = acc[u][r];
        }
    }
}

// zero-pad xproj_w (8 instances of 36x128) into 8 x 48 x 128
__global__ void pad_xproj(const float* __restrict__ w, float* __restrict__ wp)
{
    int idx = blockIdx.x * blockDim.x + threadIdx.x;
    if (idx >= 8 * XPL * DI) return;
    int k  = idx % DI;
    int n  = (idx / DI) % XPL;
    int li = idx / (DI * XPL);
    wp[idx] = (n < XPN) ? w[((size_t)li * XPN + n) * DI + k] : 0.0f;
}

// h[b,t,d] = x[b, d, t(or flipped)]  from x viewed as (B, 64, 2048)
__global__ void init_h(const float* __restrict__ x, float* __restrict__ h, int flip)
{
    int idx = blockIdx.x * blockDim.x + threadIdx.x;
    if (idx >= B_ * T_ * DM) return;
    int d = idx % DM;
    int t = (idx / DM) % T_;
    int b = idx / (DM * T_);
    int ts = flip ? (T_ - 1 - t) : t;
    h[idx] = x[((size_t)b * DM + d) * T_ + ts];
}

// mode 0: r = h, resid = r;  mode 1: r = h + resid, resid = r;  mode 2: r = h + resid (no write)
// out = LN(r, w, b).  One 64-thread block per row.
__global__ void ln_fused(const float* __restrict__ h, float* __restrict__ resid,
                         float* __restrict__ out, const float* __restrict__ w,
                         const float* __restrict__ bb, int mode)
{
    __shared__ float red[DM];
    int row = blockIdx.x;
    int j = threadIdx.x;
    size_t base = (size_t)row * DM + j;
    float r = h[base];
    if (mode != 0) r += resid[base];
    if (mode != 2) resid[base] = r;
    red[j] = r; __syncthreads();
    for (int off = 32; off > 0; off >>= 1) { if (j < off) red[j] += red[j + off]; __syncthreads(); }
    float m = red[0] * (1.0f / DM);
    __syncthreads();
    float dv = r - m;
    red[j] = dv * dv; __syncthreads();
    for (int off = 32; off > 0; off >>= 1) { if (j < off) red[j] += red[j + off]; __syncthreads(); }
    float v = red[0] * (1.0f / DM);
    out[base] = dv * rsqrtf(v + 1e-5f) * w[j] + bb[j];
}

// causal depthwise conv(4) + bias + SiLU on x = xz[..., :128]
__global__ void conv_silu(const float* __restrict__ xz, const float* __restrict__ cw,
                          const float* __restrict__ cb, float* __restrict__ xc)
{
    int idx = blockIdx.x * blockDim.x + threadIdx.x;
    if (idx >= B_ * T_ * DI) return;
    int d = idx % DI;
    int t = (idx / DI) % T_;
    int b = idx / (DI * T_);
    float acc = cb[d];
    #pragma unroll
    for (int k = 0; k < 4; ++k) {
        int tt = t - 3 + k;
        if (tt >= 0) acc += cw[d * 4 + k] * xz[((size_t)b * T_ + tt) * (2 * DI) + d];
    }
    xc[idx] = acc * sigmoidf_(acc);
}

// dt = softplus(dbl[..., :4] @ dt_w^T + dt_b)
__global__ void dt_softplus(const float* __restrict__ dbl, const float* __restrict__ dtw,
                            const float* __restrict__ dtb, float* __restrict__ dt)
{
    int idx = blockIdx.x * blockDim.x + threadIdx.x;
    if (idx >= B_ * T_ * DI) return;
    int d = idx % DI;
    int bt = idx / DI;
    float acc = dtb[d];
    #pragma unroll
    for (int r = 0; r < 4; ++r) acc += dbl[(size_t)bt * XPL + r] * dtw[d * 4 + r];
    dt[idx] = (acc > 20.0f) ? acc : log1pf(__expf(acc));
}

// selective scan: one block per batch, one thread per d_inner channel, 16 states in regs.
__global__ void ssm_scan(const float* __restrict__ xz, const float* __restrict__ xc,
                         const float* __restrict__ dbl, const float* __restrict__ dt,
                         const float* __restrict__ A_log, const float* __restrict__ Dp,
                         float* __restrict__ yb)
{
    __shared__ float sB[DS_], sC[DS_];
    int b = blockIdx.x, d = threadIdx.x;
    float An[DS_], st[DS_];
    #pragma unroll
    for (int s = 0; s < DS_; ++s) { An[s] = -__expf(A_log[d * DS_ + s]); st[s] = 0.0f; }
    float Dv = Dp[d];
    for (int t = 0; t < T_; ++t) {
        size_t bt = (size_t)b * T_ + t;
        __syncthreads();
        if (d < DS_)          sB[d]       = dbl[bt * XPL + 4 + d];        // cols 4..19
        else if (d < 2 * DS_) sC[d - DS_] = dbl[bt * XPL + 4 + d];        // cols 20..35
        __syncthreads();
        float xv  = xc[bt * DI + d];
        float dtv = dt[bt * DI + d];
        float du  = dtv * xv;
        float y = 0.0f;
        #pragma unroll
        for (int s = 0; s < DS_; ++s) {
            float dA = __expf(dtv * An[s]);
            st[s] = dA * st[s] + du * sB[s];
            y += st[s] * sC[s];
        }
        float zv = xz[bt * (2 * DI) + DI + d];
        yb[bt * DI + d] = (y + xv * Dv) * (zv * sigmoidf_(zv));
    }
}

// softmax-attention pooling over T; writes hcat[b, part*64 + d]
__global__ void attn_pool(const float* __restrict__ hs, const float* __restrict__ pw,
                          const float* __restrict__ pb, float* __restrict__ hcat, int part)
{
    __shared__ float a[T_];
    __shared__ float red[256];
    int b = blockIdx.x, tid = threadIdx.x;
    float lmax = -1e30f;
    for (int t = tid; t < T_; t += 256) {
        const float* hp = hs + ((size_t)b * T_ + t) * DM;
        float l = pb[0];
        for (int j = 0; j < DM; ++j) l += hp[j] * pw[j];
        a[t] = l;
        lmax = fmaxf(lmax, l);
    }
    red[tid] = lmax; __syncthreads();
    for (int off = 128; off > 0; off >>= 1) { if (tid < off) red[tid] = fmaxf(red[tid], red[tid + off]); __syncthreads(); }
    float mx = red[0]; __syncthreads();
    float lsum = 0.0f;
    for (int t = tid; t < T_; t += 256) { float e = __expf(a[t] - mx); a[t] = e; lsum += e; }
    red[tid] = lsum; __syncthreads();
    for (int off = 128; off > 0; off >>= 1) { if (tid < off) red[tid] += red[tid + off]; __syncthreads(); }
    float S = red[0]; __syncthreads();
    int d = tid & 63, c = tid >> 6;
    float acc = 0.0f;
    for (int t = c; t < T_; t += 4) acc += a[t] * hs[((size_t)b * T_ + t) * DM + d];
    red[tid] = acc; __syncthreads();
    if (c == 0) {
        float tot = red[d] + red[64 + d] + red[128 + d] + red[192 + d];
        hcat[b * 128 + part * 64 + d] = tot / S;
    }
}

__global__ void final_linear(const float* __restrict__ hcat, const float* __restrict__ lw,
                             const float* __restrict__ lb, float* __restrict__ out)
{
    int idx = blockIdx.x * blockDim.x + threadIdx.x;
    if (idx >= B_ * 64) return;
    int o = idx % 64, b = idx / 64;
    float acc = lb[o];
    #pragma unroll 4
    for (int j = 0; j < 128; ++j) acc += hcat[b * 128 + j] * lw[o * 128 + j];
    out[idx] = acc;
}

extern "C" void kernel_launch(void* const* d_in, const int* in_sizes, int n_in,
                              void* d_out, int out_size, void* d_ws, size_t ws_size,
                              hipStream_t stream)
{
    (void)in_sizes; (void)n_in; (void)out_size; (void)ws_size;
    const float* x       = (const float*)d_in[0];
    const float* in_w    = (const float*)d_in[1];
    const float* conv_w  = (const float*)d_in[2];
    const float* conv_b  = (const float*)d_in[3];
    const float* xproj_w = (const float*)d_in[4];
    const float* dt_w    = (const float*)d_in[5];
    const float* dt_b    = (const float*)d_in[6];
    const float* A_log   = (const float*)d_in[7];
    const float* Dmat    = (const float*)d_in[8];
    const float* out_w   = (const float*)d_in[9];
    const float* nw      = (const float*)d_in[10];
    const float* nb      = (const float*)d_in[11];
    const float* nf_w    = (const float*)d_in[12];
    const float* nf_b    = (const float*)d_in[13];
    const float* fp_w    = (const float*)d_in[14];
    const float* fp_b    = (const float*)d_in[15];
    const float* bp_w    = (const float*)d_in[16];
    const float* bp_b    = (const float*)d_in[17];
    const float* ll_w    = (const float*)d_in[18];
    const float* ll_b    = (const float*)d_in[19];

    float* ws = (float*)d_ws;
    size_t off = 0;
    float* hwork = ws + off; off += (size_t)B_ * T_ * DM;
    float* resid = ws + off; off += (size_t)B_ * T_ * DM;
    float* lnout = ws + off; off += (size_t)B_ * T_ * DM;
    float* xz    = ws + off; off += (size_t)B_ * T_ * 2 * DI;
    float* xc    = ws + off; off += (size_t)B_ * T_ * DI;
    float* dbl   = ws + off; off += (size_t)B_ * T_ * XPL;
    float* dtb   = ws + off; off += (size_t)B_ * T_ * DI;
    float* yb    = ws + off; off += (size_t)B_ * T_ * DI;
    float* hst0  = ws + off; off += (size_t)B_ * T_ * DM;
    float* hst1  = ws + off; off += (size_t)B_ * T_ * DM;
    float* hcat  = ws + off; off += (size_t)B_ * 128;
    float* xpw   = ws + off; off += (size_t)8 * XPL * DI;   // padded x-proj weights
    float* hst[2] = { hst0, hst1 };

    const int M = B_ * T_;  // 16384 rows -> 1024 M-tiles

    pad_xproj<<<(8 * XPL * DI + 255) / 256, 256, 0, stream>>>(xproj_w, xpw);

    for (int dir = 0; dir < 2; ++dir) {
        init_h<<<(B_ * T_ * DM + 255) / 256, 256, 0, stream>>>(x, hwork, dir);
        for (int i = 0; i < 4; ++i) {
            int li = dir * 4 + i;
            ln_fused<<<M, DM, 0, stream>>>(hwork, resid, lnout,
                                           nw + li * DM, nb + li * DM, (i == 0) ? 0 : 1);
            // in-proj: (M,64) x (256,64)^T -> xz (M,256); 8 groups of 16x32 strips
            gemm_wmma<64, 2><<<(M / 16) * 8 / 8, 256, 0, stream>>>(
                lnout, in_w + (size_t)li * 256 * DM, xz, 8, 2 * DI);
            conv_silu<<<(M * DI + 255) / 256, 256, 0, stream>>>(
                xz, conv_w + li * DI * 4, conv_b + li * DI, xc);
            // x-proj: (M,128) x (48,128)^T(padded) -> dbl (M,48); 1 group of 16x48
            gemm_wmma<128, 3><<<(M / 16) * 1 / 8, 256, 0, stream>>>(
                xc, xpw + (size_t)li * XPL * DI, dbl, 1, XPL);
            dt_softplus<<<(M * DI + 255) / 256, 256, 0, stream>>>(
                dbl, dt_w + li * DI * 4, dt_b + li * DI, dtb);
            ssm_scan<<<B_, DI, 0, stream>>>(xz, xc, dbl, dtb,
                                            A_log + (size_t)li * DI * DS_, Dmat + li * DI, yb);
            // out-proj: (M,128) x (64,128)^T -> hwork (M,64); 2 groups of 16x32 strips
            gemm_wmma<128, 2><<<(M / 16) * 2 / 8, 256, 0, stream>>>(
                yb, out_w + (size_t)li * DM * DI, hwork, 2, DM);
        }
        ln_fused<<<M, DM, 0, stream>>>(hwork, resid, hst[dir], nf_w, nf_b, 2);
        attn_pool<<<B_, 256, 0, stream>>>(hst[dir], dir ? bp_w : fp_w,
                                          dir ? bp_b : fp_b, hcat, dir);
    }
    final_linear<<<(B_ * 64 + 255) / 256, 256, 0, stream>>>(hcat, ll_w, ll_b, (float*)d_out);
}
